// FlaxRobertaSelfAttention_84198538871294
// MI455X (gfx1250) — matrix operand order repaired and checked by reading, not compiled, not run on hardware
//
#include <hip/hip_runtime.h>
#include <hip/hip_bf16.h>

// Problem constants
#define B_  4
#define S_  1024
#define HID_ 1024
#define H_  16
#define D_  64

typedef __attribute__((ext_vector_type(16))) __bf16 v16bf;
typedef __attribute__((ext_vector_type(8)))  float  v8f;

union Frag {
    v16bf v;
    uint4 u[2];
};

__device__ inline v8f wmma_bf16(v16bf a, v16bf b, v8f c) {
    return __builtin_amdgcn_wmma_f32_16x16x32_bf16(
        /*neg_a=*/false, a, /*neg_b=*/false, b,
        /*c_mod=*/(short)0, c, /*reuse_a=*/false, /*reuse_b=*/false);
}

__device__ inline unsigned short f2bf_bits(float f) {
    __hip_bfloat16 h = __float2bfloat16(f);
    return *reinterpret_cast<unsigned short*>(&h);
}

__device__ inline v8f vzero8() {
    return (v8f){0.f, 0.f, 0.f, 0.f, 0.f, 0.f, 0.f, 0.f};
}

// ---------------------------------------------------------------------------
// Kernel 1: convert hidden states f32 -> bf16 (row-major, M=4096 x K=1024)
// ---------------------------------------------------------------------------
__global__ __launch_bounds__(256) void cvt_hidden(const float* __restrict__ hs,
                                                  __hip_bfloat16* __restrict__ hsb,
                                                  int n) {
    int i = blockIdx.x * 256 + threadIdx.x;
    if (i < n) hsb[i] = __float2bfloat16(hs[i]);
}

// ---------------------------------------------------------------------------
// Kernel 2: convert + transpose weights -> wtb[w][n][k] = W_w[k][n]  (bf16)
// ---------------------------------------------------------------------------
__global__ __launch_bounds__(256) void cvt_weights(const float* __restrict__ Wq,
                                                   const float* __restrict__ Wk,
                                                   const float* __restrict__ Wv,
                                                   __hip_bfloat16* __restrict__ wtb) {
    int i = blockIdx.x * 256 + threadIdx.x;            // < 3*1024*1024
    int w   = i >> 20;
    int rem = i & ((1 << 20) - 1);
    int n   = rem >> 10;
    int k   = rem & 1023;
    const float* W = (w == 0) ? Wq : (w == 1) ? Wk : Wv;
    wtb[i] = __float2bfloat16(W[k * 1024 + n]);
}

// ---------------------------------------------------------------------------
// Kernel 3: fused QKV projection GEMM (bf16 WMMA, f32 accum)
//   C[M=4096, N=3072] = hsb @ [Wq|Wk|Wv] + bias, epilogue scatters to
//   Qb (B,H,S,D) scaled by 1/sqrt(D), Kb (B,H,S,D), Vt (B,H,D,S)  [all bf16]
//   Block: 256 thr = 8 waves as 2(M)x4(N); wave tile 64x64; block tile 128x256.
//   Per K-step: 8 fragment loads feed 16 WMMAs (2x the reuse of round 1).
// ---------------------------------------------------------------------------
__global__ __launch_bounds__(256) void qkv_gemm(const __hip_bfloat16* __restrict__ hsb,
                                                const __hip_bfloat16* __restrict__ wtb,
                                                const float* __restrict__ bq,
                                                const float* __restrict__ bk,
                                                const float* __restrict__ bv,
                                                __hip_bfloat16* __restrict__ Qb,
                                                __hip_bfloat16* __restrict__ Kb,
                                                __hip_bfloat16* __restrict__ Vt) {
    const int tid  = threadIdx.x;
    const int wid  = tid >> 5;
    const int lane = tid & 31;
    const int l16  = lane & 15;
    const int hh   = lane >> 4;           // lane half: selects K sub-range
    const int wm   = wid & 1;
    const int wn   = wid >> 1;
    const int Mbase = blockIdx.x * 128 + wm * 64;
    const int Nbase = blockIdx.y * 256 + wn * 64;

    v8f acc[4][4];
#pragma unroll
    for (int mi = 0; mi < 4; ++mi)
#pragma unroll
        for (int ni = 0; ni < 4; ++ni)
            acc[mi][ni] = vzero8();

    for (int kc = 0; kc < HID_; kc += 32) {
        v16bf a[4], bfrag[4];
#pragma unroll
        for (int mi = 0; mi < 4; ++mi) {
            const __hip_bfloat16* p = hsb + (Mbase + 16 * mi + l16) * HID_ + kc + 8 * hh;
            Frag f;
            f.u[0] = *reinterpret_cast<const uint4*>(p);        // K = 8h + e      (e=0..7)
            f.u[1] = *reinterpret_cast<const uint4*>(p + 16);   // K = 16+8h+(e-8) (e=8..15)
            a[mi] = f.v;
        }
#pragma unroll
        for (int ni = 0; ni < 4; ++ni) {
            const __hip_bfloat16* p = wtb + (Nbase + 16 * ni + l16) * HID_ + kc + 8 * hh;
            Frag f;
            f.u[0] = *reinterpret_cast<const uint4*>(p);        // K = 8h + e (e=0..7), 16+8h+(e-8) (e=8..15)
            f.u[1] = *reinterpret_cast<const uint4*>(p + 16);
            bfrag[ni] = f.v;
        }
#pragma unroll
        for (int mi = 0; mi < 4; ++mi)
#pragma unroll
            for (int ni = 0; ni < 4; ++ni)
                acc[mi][ni] = wmma_bf16(a[mi], bfrag[ni], acc[mi][ni]);
    }

    // Epilogue: bias add + scatter into head-major layouts
#pragma unroll
    for (int mi = 0; mi < 4; ++mi) {
#pragma unroll
        for (int ni = 0; ni < 4; ++ni) {
            const int ng = Nbase + 16 * ni + l16;    // global N in [0,3072)
            const int w  = ng >> 10;                 // 0=Q 1=K 2=V
            const int np = ng & 1023;
            const int h  = np >> 6;
            const int d  = np & 63;
            const float bias = (w == 0) ? bq[np] : (w == 1) ? bk[np] : bv[np];
#pragma unroll
            for (int r = 0; r < 8; ++r) {
                const int row = Mbase + 16 * mi + r + 8 * hh;   // global M (= b*S + s)
                const int b   = row >> 10;
                const int s   = row & 1023;
                const float val = acc[mi][ni][r] + bias;
                if (w == 0) {
                    Qb[(((b * H_ + h) * S_ + s) * D_) + d] = __float2bfloat16(val * 0.125f);
                } else if (w == 1) {
                    Kb[(((b * H_ + h) * S_ + s) * D_) + d] = __float2bfloat16(val);
                } else {
                    Vt[(((b * H_ + h) * D_ + d) * S_) + s] = __float2bfloat16(val);
                }
            }
        }
    }
}

// ---------------------------------------------------------------------------
// Kernel 4: flash attention.  Block = 8 waves, each wave owns 16 query rows.
//   grid = B*H*(S/128).  Online softmax over 64-key tiles (softmax/rescale
//   VALU cost amortized over 16 WMMAs per tile), all matmuls WMMA.
// ---------------------------------------------------------------------------
__global__ __launch_bounds__(256) void flash_attn(const __hip_bfloat16* __restrict__ Qb,
                                                  const __hip_bfloat16* __restrict__ Kb,
                                                  const __hip_bfloat16* __restrict__ Vt,
                                                  const int* __restrict__ am,
                                                  const float* __restrict__ lhm,
                                                  float* __restrict__ out) {
    __shared__ __align__(16) unsigned short pbuf[8][16][64];   // per-wave 16x64 bf16 P tile

    const int tid  = threadIdx.x;
    const int wid  = tid >> 5;
    const int lane = tid & 31;
    const int l16  = lane & 15;
    const int hh   = lane >> 4;

    const int bx = blockIdx.x;
    const int qt = bx & 7;            // query tile (128 rows each)
    const int h  = (bx >> 3) & 15;
    const int b  = bx >> 7;
    const int qbase = qt * 128 + wid * 16;

    const __hip_bfloat16* Qh = Qb + (b * H_ + h) * S_ * D_;
    const __hip_bfloat16* Kh = Kb + (b * H_ + h) * S_ * D_;
    const __hip_bfloat16* Vh = Vt + (b * H_ + h) * D_ * S_;

    // Q A-fragments: rows qbase+l16, K-dim(d) chunks [0,32) and [32,64)
    v16bf qa[2];
#pragma unroll
    for (int c = 0; c < 2; ++c) {
        const __hip_bfloat16* p = Qh + (qbase + l16) * D_ + 32 * c + 8 * hh;
        Frag f;
        f.u[0] = *reinterpret_cast<const uint4*>(p);
        f.u[1] = *reinterpret_cast<const uint4*>(p + 16);
        qa[c] = f.v;
    }

    float m_run[8], l_run[8];
    v8f o[4];
#pragma unroll
    for (int r = 0; r < 8; ++r) { m_run[r] = -1e30f; l_run[r] = 0.f; }
#pragma unroll
    for (int d2 = 0; d2 < 4; ++d2) o[d2] = vzero8();

    const float lhm_h = lhm[h];

    for (int jt = 0; jt < S_ / 64; ++jt) {
        const int k0 = jt * 64;

        // scores: 16 x 64 tile (four 16x16 C tiles); K/V fragments transient
        v8f s[4];
#pragma unroll
        for (int t = 0; t < 4; ++t) {
            const __hip_bfloat16* p = Kh + (k0 + 16 * t + l16) * D_ + 8 * hh;
            Frag f0, f1;
            f0.u[0] = *reinterpret_cast<const uint4*>(p);        // d chunk [0,32)
            f0.u[1] = *reinterpret_cast<const uint4*>(p + 16);
            f1.u[0] = *reinterpret_cast<const uint4*>(p + 32);   // d chunk [32,64)
            f1.u[1] = *reinterpret_cast<const uint4*>(p + 48);
            v8f z = vzero8();
            z    = wmma_bf16(qa[0], f0.v, z);
            s[t] = wmma_bf16(qa[1], f1.v, z);
        }

        // attention-mask bias (column-dependent only)
#pragma unroll
        for (int t = 0; t < 4; ++t) {
            const float bias = (am[b * S_ + k0 + 16 * t + l16] > 0) ? 0.f : -1e30f;
#pragma unroll
            for (int r = 0; r < 8; ++r)
                s[t][r] += bias;
        }

        // online softmax: row max across 4 tiles + 16-lane butterflies
        float mnew[8], scale[8];
#pragma unroll
        for (int r = 0; r < 8; ++r) {
            float v = fmaxf(fmaxf(s[0][r], s[1][r]), fmaxf(s[2][r], s[3][r]));
            v = fmaxf(v, __shfl_xor(v, 8, 16));
            v = fmaxf(v, __shfl_xor(v, 4, 16));
            v = fmaxf(v, __shfl_xor(v, 2, 16));
            v = fmaxf(v, __shfl_xor(v, 1, 16));
            mnew[r]  = fmaxf(m_run[r], v);
            scale[r] = __expf(m_run[r] - mnew[r]);
        }

        // exp, accumulate row sums, and spill P (bf16) to wave-private LDS
        float psum[8];
#pragma unroll
        for (int r = 0; r < 8; ++r) psum[r] = 0.f;
#pragma unroll
        for (int t = 0; t < 4; ++t)
#pragma unroll
            for (int r = 0; r < 8; ++r) {
                const float p = __expf(s[t][r] - mnew[r]);
                psum[r] += p;
                pbuf[wid][r + 8 * hh][16 * t + l16] = f2bf_bits(p);
            }
#pragma unroll
        for (int r = 0; r < 8; ++r) {
            float ps = psum[r];
            ps += __shfl_xor(ps, 8, 16);
            ps += __shfl_xor(ps, 4, 16);
            ps += __shfl_xor(ps, 2, 16);
            ps += __shfl_xor(ps, 1, 16);
            l_run[r] = l_run[r] * scale[r] + ps;
            m_run[r] = mnew[r];
        }
#pragma unroll
        for (int d2 = 0; d2 < 4; ++d2)
#pragma unroll
            for (int r = 0; r < 8; ++r)
                o[d2][r] *= scale[r];

        asm volatile("" ::: "memory");     // DS ops are in-order within a wave
        // C-layout P -> A-layout: two 16x32 A-fragments (keys [0,32),[32,64))
        Frag pf[2];
        const unsigned short* prow = &pbuf[wid][l16][0];
#pragma unroll
        for (int c = 0; c < 2; ++c) {
            pf[c].u[0] = *reinterpret_cast<const uint4*>(prow + 32 * c + 8 * hh);
            pf[c].u[1] = *reinterpret_cast<const uint4*>(prow + 32 * c + 16 + 8 * hh);
        }
        asm volatile("" ::: "memory");

        // O += P @ V  (V^T fragments are contiguous 32B loads, transient)
#pragma unroll
        for (int d2 = 0; d2 < 4; ++d2) {
            const __hip_bfloat16* p = Vh + (16 * d2 + l16) * S_ + k0 + 8 * hh;
            Frag f0, f1;
            f0.u[0] = *reinterpret_cast<const uint4*>(p);        // keys [0,32)
            f0.u[1] = *reinterpret_cast<const uint4*>(p + 16);
            f1.u[0] = *reinterpret_cast<const uint4*>(p + 32);   // keys [32,64)
            f1.u[1] = *reinterpret_cast<const uint4*>(p + 48);
            o[d2] = wmma_bf16(pf[0].v, f0.v, o[d2]);
            o[d2] = wmma_bf16(pf[1].v, f1.v, o[d2]);
        }
    }

    // epilogue: head mask * softmax normalization, write (B,S,HID) f32
#pragma unroll
    for (int r = 0; r < 8; ++r) {
        const float inv = lhm_h / l_run[r];
        const int qrow = qbase + r + 8 * hh;
#pragma unroll
        for (int d2 = 0; d2 < 4; ++d2) {
            out[(b * S_ + qrow) * HID_ + h * D_ + 16 * d2 + l16] = o[d2][r] * inv;
        }
    }
}

// ---------------------------------------------------------------------------
// Host-side launcher
// ---------------------------------------------------------------------------
extern "C" void kernel_launch(void* const* d_in, const int* in_sizes, int n_in,
                              void* d_out, int out_size, void* d_ws, size_t ws_size,
                              hipStream_t stream) {
    const float* hs  = (const float*)d_in[0];
    const int*   am  = (const int*)  d_in[1];
    const float* lhm = (const float*)d_in[2];
    const float* Wq  = (const float*)d_in[3];
    const float* bq  = (const float*)d_in[4];
    const float* Wk  = (const float*)d_in[5];
    const float* bk  = (const float*)d_in[6];
    const float* Wv  = (const float*)d_in[7];
    const float* bv  = (const float*)d_in[8];
    float* out = (float*)d_out;

    char* ws = (char*)d_ws;
    __hip_bfloat16* hsb = (__hip_bfloat16*)(ws);                          // 8 MB
    __hip_bfloat16* wtb = (__hip_bfloat16*)(ws + ((size_t)8  << 20));     // 6 MB
    __hip_bfloat16* Qb  = (__hip_bfloat16*)(ws + ((size_t)14 << 20));     // 8 MB
    __hip_bfloat16* Kb  = (__hip_bfloat16*)(ws + ((size_t)22 << 20));     // 8 MB
    __hip_bfloat16* Vt  = (__hip_bfloat16*)(ws + ((size_t)30 << 20));     // 8 MB

    const int n_hidden = B_ * S_ * HID_;          // 4 Mi elements
    cvt_hidden<<<n_hidden / 256, 256, 0, stream>>>(hs, hsb, n_hidden);
    cvt_weights<<<(3 * HID_ * HID_) / 256, 256, 0, stream>>>(Wq, Wk, Wv, wtb);
    qkv_gemm<<<dim3(32, 12), 256, 0, stream>>>(hsb, wtb, bq, bk, bv, Qb, Kb, Vt);
    flash_attn<<<B_ * H_ * (S_ / 128), 256, 0, stream>>>(Qb, Kb, Vt, am, lhm, out);
}
